// TransformerModel_Hetero4_11716670783749
// MI455X (gfx1250) — matrix operand order repaired and checked by reading, not compiled, not run on hardware
//
#include <hip/hip_runtime.h>
#include <cmath>

// ---------------------------------------------------------------------------
// TransformerModel_Hetero4 for MI455X (gfx1250, wave32, WMMA + async-to-LDS).
//
// v3:
//  - Non-gather GEMMs stage A/B tiles with GLOBAL_LOAD_ASYNC_TO_LDS_B128
//    (inline asm; ASYNCcnt-tracked DMA, no VGPR round trip), 4-deep LDS
//    pipeline, one barrier per K-step, s_wait_asynccnt threshold waits.
//  - Gather GEMM (reg layer 1, hcat) keeps register staging but is also
//    double-buffered with a single barrier per K-step.
//  - All GEMM operands bf16 at rest (weights pre-transposed/padded per
//    launch; activation producers emit bf16).
// Dense math: v_wmma_f32_16x16x32_bf16, fp32 accumulate.
// ---------------------------------------------------------------------------

#define N_NODES 10000
#define NPAD    10048L   // 157 * 64 (grid extent for node GEMMs)
#define E_CONN  131072
#define E_OD    65536
#define NH      8
#define OH      512      // O_HEAD

typedef __attribute__((ext_vector_type(16))) __bf16         v16bf;
typedef __attribute__((ext_vector_type(8)))  float          v8f;
typedef __attribute__((ext_vector_type(8)))  unsigned short us8;
typedef unsigned short ushort_t;

union Frag16 { us8 h[2]; v16bf v; };

__device__ __forceinline__ ushort_t f2bf(float f) {
  union { float f; unsigned u; } x; x.f = f;
  unsigned r = x.u + 0x7FFFu + ((x.u >> 16) & 1u);   // round-to-nearest-even
  return (ushort_t)(r >> 16);
}
__device__ __forceinline__ float bf2f(ushort_t u) {
  union { unsigned u; float f; } x; x.u = ((unsigned)u) << 16; return x.f;
}

// Async DMA: 16B per lane, global -> LDS, tracked by ASYNCcnt.
__device__ __forceinline__ void async_b128(unsigned ldsOff, const ushort_t* g) {
  asm volatile("global_load_async_to_lds_b128 %0, %1, off"
               :: "v"(ldsOff), "v"(g) : "memory");
}

struct GatherArgs {
  const int*      src;
  const int*      dst;
  const ushort_t* h;      // [N, 512] bf16
  const float*    coord;  // [N, 2]
  const float*    ef;     // [E, 2]
};

// ---------------------------------------------------------------------------
// WMMA GEMM:  C[M,Nout] = act( A_bf16[M,Kpad] @ W_bf16[Kpad,Nout] + bias (+add) )
// W stored transposed+padded: Wt[Nout][Kpad]. 64x64 block tile, 8 waves,
// 2 accumulators/wave, 4-deep async-to-LDS pipeline (non-gather path).
// ---------------------------------------------------------------------------
template<bool GATHER, bool RELU, bool ADDEND, bool OUTBF>
__global__ __launch_bounds__(256) void gemm_wmma(
    const ushort_t* __restrict__ A, int lda, int M, int Kpad,
    const ushort_t* __restrict__ Wt,     // [Nout][Kpad] bf16
    int Nout,
    const float* __restrict__ bias,
    const float* __restrict__ add,
    void* __restrict__ Cout,
    GatherArgs ga)
{
  constexpr int DEPTH = 4;               // async pipeline slots (power of 2)
  __shared__ ushort_t As[DEPTH][64][32]; // [slot][m][k]
  __shared__ ushort_t Bs[DEPTH][64][32]; // [slot][n][k]

  const int tid  = threadIdx.x;
  const int wave = tid >> 5;
  const int lane = tid & 31;
  const int wm   = wave & 3;       // M sub-tile (0..3)
  const int wn   = wave >> 2;      // N sub-tile (0..1, 32 cols)
  const int half = lane >> 4;
  const int l16  = lane & 15;
  const int bM   = blockIdx.y * 64;
  const int bN   = blockIdx.x * 64;

  v8f acc0 = {}, acc1 = {};

  const int srow = tid >> 2;        // 0..63: A row / B col within tile
  const int sk   = (tid & 3) * 8;   // 0,8,16,24

  const ushort_t* aRow = nullptr;
  int gsrc = 0, gdst = 0;
  if constexpr (!GATHER) {
    aRow = A + (long)(bM + srow) * lda;
  } else {
    gsrc = ga.src[bM + srow];
    gdst = ga.dst[bM + srow];
  }
  const ushort_t* bRow = Wt + (long)(bN + srow) * Kpad;

  const int nT = Kpad >> 5;                   // number of 32-wide K tiles

  // gather-mode register staging
  us8 rA = {}, rB = {};
  auto fetchA = [&](int kk) -> us8 {
    if (kk < 512)  return *(const us8*)(ga.h + (long)gsrc * OH + kk);
    if (kk < 1024) return *(const us8*)(ga.h + (long)gdst * OH + (kk - 512));
    us8 r;
    const int e = bM + srow;
    #pragma unroll
    for (int j = 0; j < 8; ++j) {
      const int c = kk + j;
      float v = 0.f;
      if (c < 1026)      v = ga.coord[(long)gsrc * 2 + (c - 1024)];
      else if (c < 1028) v = ga.coord[(long)gdst * 2 + (c - 1026)];
      else if (c < 1030) v = ga.ef[(long)e * 2 + (c - 1028)];
      r[j] = f2bf(v);
    }
    return r;
  };

  if constexpr (!GATHER) {
    // prologue: issue tiles 0..DEPTH-2
    #pragma unroll
    for (int t = 0; t < DEPTH - 1; ++t) {
      const int tt = (t < nT) ? t : (nT - 1);             // clamp (redundant ok)
      async_b128((unsigned)(size_t)&As[t][srow][sk], aRow + (tt << 5) + sk);
      async_b128((unsigned)(size_t)&Bs[t][srow][sk], bRow + (tt << 5) + sk);
    }
  } else {
    rA = fetchA(sk);
    rB = *(const us8*)(bRow + sk);
  }

  for (int t = 0; t < nT; ++t) {
    int buf;
    if constexpr (!GATHER) {
      buf = t & (DEPTH - 1);
      // in-order completion: <=4 outstanding ==> oldest tile (t) has landed
      asm volatile("s_wait_asynccnt 0x4" ::: "memory");
      __syncthreads();   // whole tile t visible; all waves done with tile t-1
      {
        // look-ahead issue (clamped past the end; dummy slots never read).
        const int tn = t + DEPTH - 1;
        const int tt = (tn < nT) ? tn : (nT - 1);
        const int slot = tn & (DEPTH - 1);
        async_b128((unsigned)(size_t)&As[slot][srow][sk], aRow + (tt << 5) + sk);
        async_b128((unsigned)(size_t)&Bs[slot][srow][sk], bRow + (tt << 5) + sk);
      }
    } else {
      buf = t & 1;
      *(us8*)&As[buf][srow][sk] = rA;       // slot read by compute(t-2): done pre-barrier(t-1)
      *(us8*)&Bs[buf][srow][sk] = rB;
      __syncthreads();                       // compiler inserts dscnt wait
      if (t + 1 < nT) {                      // prefetch overlaps compute
        rA = fetchA(((t + 1) << 5) + sk);
        rB = *(const us8*)(bRow + ((t + 1) << 5) + sk);
      }
    }

    // A fragment: lane(m=l16, half): K = half*8+{0..7}, 16+half*8+{0..7}
    Frag16 fa;
    {
      const int am = wm * 16 + l16;
      fa.h[0] = *(const us8*)&As[buf][am][half * 8];
      fa.h[1] = *(const us8*)&As[buf][am][16 + half * 8];
    }
    // B fragments: lane(n=l16, half): K = half*8+{0..7}, 16+half*8+{0..7}
    Frag16 fb0, fb1;
    {
      const int n0 = wn * 32 + l16;
      fb0.h[0] = *(const us8*)&Bs[buf][n0][half * 8];
      fb0.h[1] = *(const us8*)&Bs[buf][n0][16 + half * 8];
      fb1.h[0] = *(const us8*)&Bs[buf][n0 + 16][half * 8];
      fb1.h[1] = *(const us8*)&Bs[buf][n0 + 16][16 + half * 8];
    }

    acc0 = __builtin_amdgcn_wmma_f32_16x16x32_bf16(
        false, fa.v, false, fb0.v, (short)0, acc0, false, false);
    acc1 = __builtin_amdgcn_wmma_f32_16x16x32_bf16(
        false, fa.v, false, fb1.v, (short)0, acc1, false, false);
  }

  // ---- epilogue: C layout m = r + 8*half, n = l16 ----
  const int cmBase = bM + wm * 16 + half * 8;
  float*    Cf = (float*)Cout;
  ushort_t* Ch = (ushort_t*)Cout;
  #pragma unroll
  for (int r = 0; r < 8; ++r) {
    const int gm = cmBase + r;
    if (gm >= M) continue;
    #pragma unroll
    for (int sub = 0; sub < 2; ++sub) {
      const int gn = bN + wn * 32 + sub * 16 + l16;
      float v = (sub ? acc1[r] : acc0[r]) + bias[gn];
      if (ADDEND) v += add[(long)gm * Nout + gn];
      if (RELU)   v = fmaxf(v, 0.f);
      if (OUTBF) Ch[(long)gm * Nout + gn] = f2bf(v);
      else       Cf[(long)gm * Nout + gn] = v;
    }
  }
}

// ---------------------------------------------------------------------------
// Weight convert: dst[Nout][Kpad] (bf16) = transpose(src[Ka][Nout]), 0-padded.
// ---------------------------------------------------------------------------
__global__ void cvt_wT(ushort_t* __restrict__ dst, const float* __restrict__ src,
                       int Ka, int Kpad, int Nout, long total) {
  const long i = (long)blockIdx.x * blockDim.x + threadIdx.x;
  if (i >= total) return;
  const int k = (int)(i / Nout);          // reads coalesced along n
  const int n = (int)(i % Nout);
  const float v = (k < Ka) ? src[(long)k * Nout + n] : 0.f;
  dst[(long)n * Kpad + k] = f2bf(v);
}

__global__ void fill0(float* __restrict__ p, long n) {
  long i  = (long)blockIdx.x * blockDim.x + threadIdx.x;
  long st = (long)gridDim.x * blockDim.x;
  for (; i < n; i += st) p[i] = 0.f;
}

// node MLP: relu(x@w1+b1)@w2+b2 ; concat coord; pad to 96 -> bf16 x66 [N,96]
__global__ void pre_k(const float* __restrict__ nf, const float* __restrict__ coord,
                      const float* __restrict__ w1, const float* __restrict__ b1,
                      const float* __restrict__ w2, const float* __restrict__ b2,
                      ushort_t* __restrict__ x66, int N) {
  __shared__ float t[64];
  const int n = blockIdx.x;
  if (n >= N) return;
  const int j = threadIdx.x;
  float a = b1[j];
  #pragma unroll
  for (int i = 0; i < 16; ++i) a += nf[n * 16 + i] * w1[i * 64 + j];
  t[j] = fmaxf(a, 0.f);
  __syncthreads();
  float c = b2[j];
  for (int i = 0; i < 64; ++i) c += t[i] * w2[i * 64 + j];
  x66[(long)n * 96 + j] = f2bf(c);
  if (j < 2)  x66[(long)n * 96 + 64 + j] = f2bf(coord[n * 2 + j]);
  if (j < 30) x66[(long)n * 96 + 66 + j] = 0;   // K padding
}

// wave-per-edge score: per head exp(clip((k[src].q[dst])*ef/sqrt, -5, 5));
// accumulates z[dst,h] with fp32 atomics.
__global__ void score_k(const float* __restrict__ Q, const float* __restrict__ K,
                        const int* __restrict__ src, const int* __restrict__ dst,
                        const float* __restrict__ ef, float scale,
                        float* __restrict__ sc, float* __restrict__ z, int E) {
  const int e = blockIdx.x * 8 + (threadIdx.x >> 5);
  if (e >= E) return;
  const int lane = threadIdx.x & 31;
  const float* kp = K + (long)src[e] * OH;
  const float* qp = Q + (long)dst[e] * OH;
  const float m = ef ? ef[(long)e * 2] : 1.f;
  #pragma unroll
  for (int h = 0; h < NH; ++h) {
    const int c = h * 64 + lane;
    float a = kp[c] * qp[c] + kp[c + 32] * qp[c + 32];
    #pragma unroll
    for (int o = 16; o; o >>= 1) a += __shfl_xor(a, o, 32);
    if (lane == 0) {
      float t = fminf(5.f, fmaxf(-5.f, a * m * scale));
      t = __expf(t);
      sc[(long)e * NH + h] = t;
      unsafeAtomicAdd(&z[(long)dst[e] * NH + h], t);
    }
  }
}

// wv[dst, c] += v[src, c] * score[e, c/64]
__global__ void scatter_wv(const float* __restrict__ V, const int* __restrict__ src,
                           const int* __restrict__ dst, const float* __restrict__ sc,
                           float* __restrict__ wv, long total) {
  const long i = (long)blockIdx.x * blockDim.x + threadIdx.x;
  if (i >= total) return;
  const long e = i >> 9;
  const int  c = (int)(i & 511);
  const float val = V[(long)src[e] * OH + c] * sc[e * NH + (c >> 6)];
  unsafeAtomicAdd(&wv[(long)dst[e] * OH + c], val);
}

// out_bf16 = wv / (z + addone)
__global__ void wv_norm(const float* __restrict__ wv, const float* __restrict__ z,
                        float addone, ushort_t* __restrict__ out, long n512) {
  const long i = (long)blockIdx.x * blockDim.x + threadIdx.x;
  if (i >= n512) return;
  const long row = i >> 9;
  const int  c   = (int)(i & 511);
  out[i] = f2bf(wv[i] / (z[row * NH + (c >> 6)] + addone));
}

// out_bf16 = leaky_relu(h + layernorm(h)*g + b), wave per 512-row
__global__ void ln_leaky(const float* __restrict__ h, const float* __restrict__ g,
                         const float* __restrict__ bb, ushort_t* __restrict__ out, int N) {
  const int row = blockIdx.x * 8 + (threadIdx.x >> 5);
  if (row >= N) return;
  const int lane = threadIdx.x & 31;
  float v[16];
  float s = 0.f;
  #pragma unroll
  for (int i = 0; i < 16; ++i) {
    v[i] = h[(long)row * OH + lane + 32 * i];
    s += v[i];
  }
  #pragma unroll
  for (int o = 16; o; o >>= 1) s += __shfl_xor(s, o, 32);
  const float mean = s * (1.f / 512.f);
  float vs = 0.f;
  #pragma unroll
  for (int i = 0; i < 16; ++i) { const float d = v[i] - mean; vs += d * d; }
  #pragma unroll
  for (int o = 16; o; o >>= 1) vs += __shfl_xor(vs, o, 32);
  const float rs = rsqrtf(vs * (1.f / 512.f) + 1e-5f);
  #pragma unroll
  for (int i = 0; i < 16; ++i) {
    const int c = lane + 32 * i;
    float y = v[i] + ((v[i] - mean) * rs * g[c] + bb[c]);
    y = (y > 0.f) ? y : 0.01f * y;           // jax.nn.leaky_relu default
    out[(long)row * OH + c] = f2bf(y);
  }
}

// out[e] = bf16 r[e,:] . w + b   (wave per edge, 128 cols)
__global__ void final_dot(const ushort_t* __restrict__ r, const float* __restrict__ w,
                          const float* __restrict__ b, float* __restrict__ out, int E) {
  const int e = blockIdx.x * 8 + (threadIdx.x >> 5);
  if (e >= E) return;
  const int lane = threadIdx.x & 31;
  float acc = 0.f;
  #pragma unroll
  for (int i = 0; i < 4; ++i) {
    const int c = lane + 32 * i;
    acc += bf2f(r[(long)e * 128 + c]) * w[c];
  }
  #pragma unroll
  for (int o = 16; o; o >>= 1) acc += __shfl_xor(acc, o, 32);
  if (lane == 0) out[e] = acc + b[0];
}

// ---------------------------------------------------------------------------
struct MhaW { const ushort_t *wqT, *wkT, *wvT, *woT, *wfT; };

// One MHA layer. f0/f1/f2 are f32 node temps, result bf16 -> outbf.
//   q=f0 k=f1 v=f2 ; wv=f0 (q dead) ; o=f1 (k dead) ; hnew=f2 (v dead)
static inline void run_mha(hipStream_t s,
                           const ushort_t* xbf, int lda, int Kpad,
                           const MhaW& W, const float* const* pp,  // bias/ln: pp[1,3,5,7,9,10,11]
                           const int* src, const int* dst, int E,
                           const float* ef, float scale, float addone,
                           float* f0, float* f1, float* f2,
                           ushort_t* wvbf, ushort_t* outbf,
                           float* sbuf, float* zbuf) {
  const dim3 gN(OH / 64, (int)(NPAD / 64));
  GatherArgs none{};
  gemm_wmma<false, false, false, false><<<gN, 256, 0, s>>>(xbf, lda, N_NODES, Kpad, W.wqT, OH, pp[1], nullptr, f0, none);
  gemm_wmma<false, false, false, false><<<gN, 256, 0, s>>>(xbf, lda, N_NODES, Kpad, W.wkT, OH, pp[3], nullptr, f1, none);
  gemm_wmma<false, false, false, false><<<gN, 256, 0, s>>>(xbf, lda, N_NODES, Kpad, W.wvT, OH, pp[5], nullptr, f2, none);
  fill0<<<512, 256, 0, s>>>(zbuf, (long)N_NODES * NH);
  score_k<<<(E + 7) / 8, 256, 0, s>>>(f0, f1, src, dst, ef, scale, sbuf, zbuf, E);
  fill0<<<4096, 256, 0, s>>>(f0, (long)N_NODES * OH);                   // f0 -> wv accumulator
  const long tot = (long)E * OH;
  scatter_wv<<<(int)((tot + 255) / 256), 256, 0, s>>>(f2, src, dst, sbuf, f0, tot);
  wv_norm<<<(int)(((long)N_NODES * OH + 255) / 256), 256, 0, s>>>(f0, zbuf, addone, wvbf, (long)N_NODES * OH);
  gemm_wmma<false, false, false, false><<<gN, 256, 0, s>>>(wvbf, OH, N_NODES, OH, W.woT, OH, pp[7], nullptr, f1, none); // o
  gemm_wmma<false, false, true,  false><<<gN, 256, 0, s>>>(xbf, lda, N_NODES, Kpad, W.wfT, OH, pp[9], f1, f2, none);   // h
  ln_leaky<<<(N_NODES + 7) / 8, 256, 0, s>>>(f2, pp[10], pp[11], outbf, N_NODES);
}

// ---------------------------------------------------------------------------
// Entry. Param flattening: recursive insertion order of setup_inputs():
//   P[0..3] pre, P[4..15] conv1 {wq,bq,wk,bk,wv,bv,wo,bo,wf,bf,ln_g,ln_b},
//   P[16..27] cc1, P[28..39] cc2, P[40..44] reg_w, P[45..49] reg_b
// ---------------------------------------------------------------------------
extern "C" void kernel_launch(void* const* d_in, const int* in_sizes, int n_in,
                              void* d_out, int out_size, void* d_ws, size_t ws_size,
                              hipStream_t stream) {
  (void)in_sizes; (void)n_in; (void)out_size; (void)ws_size;

  const float* node_feat = (const float*)d_in[0];
  const float* edge_feat = (const float*)d_in[1];
  const float* coord     = (const float*)d_in[2];
  const int*   od_src    = (const int*)d_in[3];
  const int*   od_dst    = (const int*)d_in[4];
  const int*   conn_src  = (const int*)d_in[5];
  const int*   conn_dst  = (const int*)d_in[6];
  const float* const* P  = (const float* const*)(d_in + 7);

  // ---- workspace carve: f32 region then bf16 region ----
  float* f0   = (float*)d_ws;
  float* f1   = f0 + NPAD * OH;
  float* f2   = f1 + NPAD * OH;
  float* sbuf = f2 + NPAD * OH;            // E_CONN*8 (covers E_OD too)
  float* zbuf = sbuf + (long)E_CONN * NH;  // NPAD*8

  ushort_t* cur = (ushort_t*)(zbuf + NPAD * NH);
  auto take = [&](long n) { ushort_t* p = cur; cur += n; return p; };
  ushort_t* x66bf = take(NPAD * 96);
  ushort_t* hA    = take(NPAD * OH);
  ushort_t* hB    = take(NPAD * OH);
  ushort_t* wvbf  = take(NPAD * OH);
  ushort_t* rbf0  = take((long)E_CONN * 128);
  ushort_t* rbf1  = take((long)E_CONN * 128);
  MhaW W1{take(512L * 96), take(512L * 96), take(512L * 96), take(512L * 512), take(512L * 96)};
  MhaW W2{take(512L * 512), take(512L * 512), take(512L * 512), take(512L * 512), take(512L * 512)};
  MhaW W3{take(512L * 512), take(512L * 512), take(512L * 512), take(512L * 512), take(512L * 512)};
  ushort_t* rw0 = take(128L * 1056);
  ushort_t* rw1 = take(128L * 128);
  ushort_t* rw2 = take(128L * 128);
  ushort_t* rw3 = take(128L * 128);

  // ---- per-launch weight conversion (transpose + pad + bf16) ----
  auto cvt = [&](const float* src, int Ka, int Kpad, int Nout, ushort_t* dst) {
    const long tot = (long)Nout * Kpad;
    cvt_wT<<<(int)((tot + 255) / 256), 256, 0, stream>>>(dst, src, Ka, Kpad, Nout, tot);
  };
  cvt(P[4], 66, 96, OH, (ushort_t*)W1.wqT);  cvt(P[6], 66, 96, OH, (ushort_t*)W1.wkT);
  cvt(P[8], 66, 96, OH, (ushort_t*)W1.wvT);  cvt(P[10], OH, OH, OH, (ushort_t*)W1.woT);
  cvt(P[12], 66, 96, OH, (ushort_t*)W1.wfT);
  cvt(P[16], OH, OH, OH, (ushort_t*)W2.wqT); cvt(P[18], OH, OH, OH, (ushort_t*)W2.wkT);
  cvt(P[20], OH, OH, OH, (ushort_t*)W2.wvT); cvt(P[22], OH, OH, OH, (ushort_t*)W2.woT);
  cvt(P[24], OH, OH, OH, (ushort_t*)W2.wfT);
  cvt(P[28], OH, OH, OH, (ushort_t*)W3.wqT); cvt(P[30], OH, OH, OH, (ushort_t*)W3.wkT);
  cvt(P[32], OH, OH, OH, (ushort_t*)W3.wvT); cvt(P[34], OH, OH, OH, (ushort_t*)W3.woT);
  cvt(P[36], OH, OH, OH, (ushort_t*)W3.wfT);
  cvt(P[40], 1030, 1056, 128, rw0);
  cvt(P[41], 128, 128, 128, rw1);
  cvt(P[42], 128, 128, 128, rw2);
  cvt(P[43], 128, 128, 128, rw3);

  // 1) node MLP + coord concat -> bf16 x66 (Kpad=96)
  pre_k<<<N_NODES, 64, 0, stream>>>(node_feat, coord, P[0], P[1], P[2], P[3], x66bf, N_NODES);

  // 2) conv1 over OD edges (in_f=66, z gets +1) -> hA
  run_mha(stream, x66bf, 96, 96, W1, P + 4, od_src, od_dst, E_OD,
          nullptr, 1.0f / sqrtf(66.0f), 1.0f, f0, f1, f2, wvbf, hA, sbuf, zbuf);
  // 3) cc1 (score *= ef[:,0]) -> hB
  run_mha(stream, hA, OH, OH, W2, P + 16, conn_src, conn_dst, E_CONN,
          edge_feat, 1.0f / sqrtf(512.0f), 0.0f, f0, f1, f2, wvbf, hB, sbuf, zbuf);
  // 4) cc2 -> hA
  run_mha(stream, hB, OH, OH, W3, P + 28, conn_src, conn_dst, E_CONN,
          edge_feat, 1.0f / sqrtf(512.0f), 0.0f, f0, f1, f2, wvbf, hA, sbuf, zbuf);

  // 5) regression head: gathered hcat(1030) -> 128 -> 128 -> 128 -> 128 -> 1
  GatherArgs ga{conn_src, conn_dst, hA, coord, edge_feat};
  GatherArgs none{};
  const dim3 gR(128 / 64, E_CONN / 64);
  gemm_wmma<true,  true, false, true><<<gR, 256, 0, stream>>>(nullptr, 0, E_CONN, 1056, rw0, 128, P[45], nullptr, rbf0, ga);
  gemm_wmma<false, true, false, true><<<gR, 256, 0, stream>>>(rbf0, 128, E_CONN, 128, rw1, 128, P[46], nullptr, rbf1, none);
  gemm_wmma<false, true, false, true><<<gR, 256, 0, stream>>>(rbf1, 128, E_CONN, 128, rw2, 128, P[47], nullptr, rbf0, none);
  gemm_wmma<false, true, false, true><<<gR, 256, 0, stream>>>(rbf0, 128, E_CONN, 128, rw3, 128, P[48], nullptr, rbf1, none);
  final_dot<<<(E_CONN + 7) / 8, 256, 0, stream>>>(rbf1, P[44], P[49], (float*)d_out, E_CONN);
}
